// EncoderLayerSelfSem_44822278701758
// MI455X (gfx1250) — matrix operand order repaired and checked by reading, not compiled, not run on hardware
//
#include <hip/hip_runtime.h>
#include <math.h>

typedef __bf16 bf16;
typedef __attribute__((ext_vector_type(4)))  bf16  v4bf;
typedef __attribute__((ext_vector_type(8)))  bf16  v8bf;
typedef __attribute__((ext_vector_type(16))) bf16  v16bf;
typedef __attribute__((ext_vector_type(8)))  float v8f;
typedef __attribute__((ext_vector_type(4)))  float v4f;

#define LDSW 40            // LDS row stride in bf16 elems (32 data + 8 pad = 80B, 16B-aligned)
#define BUFE (2 * 128 * LDSW)   // elements per double-buffer stage (A tile + B tile)

enum { EPI_BIAS = 1, EPI_BF16N = 2, EPI_BF16T = 4, EPI_SCALE = 8 };

__device__ __forceinline__ v16bf frag_cat(v8bf lo, v8bf hi) {
  v16bf r;
#pragma unroll
  for (int i = 0; i < 8; i++) { r[i] = lo[i]; r[i + 8] = hi[i]; }
  return r;
}

// CDNA5 async DMA: global -> LDS, 16B per lane, tracked by ASYNCcnt.
__device__ __forceinline__ void async_b128(unsigned lds_addr, unsigned goff,
                                           unsigned long long sbase) {
  asm volatile("global_load_async_to_lds_b128 %0, %1, %2 offset:0"
               :: "v"(lds_addr), "v"(goff), "s"(sbase) : "memory");
}

__device__ __forceinline__ void wait_async0() {
#if __has_builtin(__builtin_amdgcn_s_wait_asynccnt)
  __builtin_amdgcn_s_wait_asynccnt(0);
#else
  asm volatile("s_wait_asynccnt 0" ::: "memory");
#endif
}

// C(M x N) = A(M x K) * BT(N x K)^T ; tile dims multiples of 128 (K of 32).
// 256 threads = 8 waves; 128x128 block tile; each wave: 2x4 subtiles of 16x16.
// Double-buffered LDS staging via global_load_async_to_lds_b128.
template<int EPI>
__global__ __launch_bounds__(256)
void gemm_wmma(const bf16* __restrict__ A, const bf16* __restrict__ BT,
               void* __restrict__ Cv, const float* __restrict__ bias,
               int K, int lda, int ldbt, int ldc,
               long long aBatch, long long btBatch, long long cBatch,
               float scale)
{
  __shared__ bf16 lds[2 * BUFE];   // [buf][{A,B}][128][LDSW]
  const int tid  = threadIdx.x;
  const int wave = tid >> 5;
  const int lane = tid & 31;
  const int half = lane >> 4;
  const int l16  = lane & 15;
  A  += (long long)blockIdx.z * aBatch;
  BT += (long long)blockIdx.z * btBatch;
  const int tileM = blockIdx.y * 128;
  const int tileN = blockIdx.x * 128;
  const int wm = (wave & 3) * 32;   // wave row offset in tile
  const int wn = (wave >> 2) * 64;  // wave col offset in tile

  // Per-thread staging chunks: 512 16B chunks per 128x32 tile, 2 per thread.
  const int r0 = tid >> 2,          c0 = (tid & 3) * 8;
  const int r1 = (tid + 256) >> 2,  c1 = ((tid + 256) & 3) * 8;
  const unsigned long long baseA  = (unsigned long long)A;
  const unsigned long long baseBT = (unsigned long long)BT;
  const unsigned aOff0 = (unsigned)(((tileM + r0) * lda  + c0) * 2);
  const unsigned aOff1 = (unsigned)(((tileM + r1) * lda  + c1) * 2);
  const unsigned bOff0 = (unsigned)(((tileN + r0) * ldbt + c0) * 2);
  const unsigned bOff1 = (unsigned)(((tileN + r1) * ldbt + c1) * 2);
  // LDS byte addresses (low 32 bits of the generic AS3 pointer = LDS offset)
  const unsigned aLds0 = (unsigned)(size_t)(lds + r0 * LDSW + c0);
  const unsigned aLds1 = (unsigned)(size_t)(lds + r1 * LDSW + c1);
  const unsigned bLds0 = (unsigned)(size_t)(lds + 128 * LDSW + r0 * LDSW + c0);
  const unsigned bLds1 = (unsigned)(size_t)(lds + 128 * LDSW + r1 * LDSW + c1);

  auto issue = [&](int k0, int buf) {
    const unsigned bb = (unsigned)buf * (BUFE * 2);   // byte offset of stage
    const unsigned kb = (unsigned)(k0 * 2);
    async_b128(aLds0 + bb, aOff0 + kb, baseA);
    async_b128(aLds1 + bb, aOff1 + kb, baseA);
    async_b128(bLds0 + bb, bOff0 + kb, baseBT);
    async_b128(bLds1 + bb, bOff1 + kb, baseBT);
  };

  v8f acc[2][4];
#pragma unroll
  for (int i = 0; i < 2; i++)
#pragma unroll
    for (int j = 0; j < 4; j++) acc[i][j] = {};

  const int nIter = K >> 5;
  issue(0, 0);
  int buf = 0;
  for (int it = 0; it < nIter; ++it) {
    wait_async0();        // my async writes (into buf) complete
    __syncthreads();      // everyone's writes complete; prior reads of buf^1 done
    if (it + 1 < nIter) issue((it + 1) << 5, buf ^ 1);

    const bf16* sA = lds + buf * BUFE;
    const bf16* sB = sA + 128 * LDSW;
    v16bf aF[2], bF[4];
    // A fragment (16-bit A 16x32 layout): half 0 -> K {0..7,16..23}, half 1 -> {8..15,24..31}
#pragma unroll
    for (int i = 0; i < 2; i++) {
      const bf16* p = sA + (wm + i * 16 + l16) * LDSW + half * 8;
      aF[i] = frag_cat(*(const v8bf*)p, *(const v8bf*)(p + 16));
    }
    // B fragment: half 0 -> K {0..7,16..23}, half 1 -> {8..15,24..31} (col n = lane%16)
#pragma unroll
    for (int j = 0; j < 4; j++) {
      const bf16* p = sB + (wn + j * 16 + l16) * LDSW + half * 8;
      bF[j] = frag_cat(*(const v8bf*)p, *(const v8bf*)(p + 16));
    }
#pragma unroll
    for (int i = 0; i < 2; i++)
#pragma unroll
      for (int j = 0; j < 4; j++)
        acc[i][j] = __builtin_amdgcn_wmma_f32_16x16x32_bf16(
            false, aF[i], false, bF[j], (short)0, acc[i][j], false, false);
    buf ^= 1;
  }

  // Epilogue. C-subtile layout: lane col n = lane%16, rows m = v + 8*(lane/16).
#pragma unroll
  for (int i = 0; i < 2; i++) {
    const int mBase = tileM + wm + i * 16 + half * 8;  // rows mBase+v, v=0..7
#pragma unroll
    for (int j = 0; j < 4; j++) {
      const int nCol = tileN + wn + j * 16 + l16;
      v8f c = acc[i][j];
      if (EPI & EPI_SCALE) {
#pragma unroll
        for (int v = 0; v < 8; v++) c[v] *= scale;
      }
      if (EPI & EPI_BIAS) {
        const float bv = bias[nCol];
#pragma unroll
        for (int v = 0; v < 8; v++) c[v] += bv;
      }
      if (EPI & EPI_BF16T) {
        // transposed store: one contiguous b128 per lane
        bf16* C = (bf16*)Cv + (long long)blockIdx.z * cBatch;
        v8bf o;
#pragma unroll
        for (int v = 0; v < 8; v++) o[v] = (bf16)c[v];
        *(v8bf*)(C + (long long)nCol * ldc + mBase) = o;
      } else if (EPI & EPI_BF16N) {
        bf16* C = (bf16*)Cv + (long long)blockIdx.z * cBatch;
#pragma unroll
        for (int v = 0; v < 8; v++)
          C[(long long)(mBase + v) * ldc + nCol] = (bf16)c[v];
      } else {
        float* C = (float*)Cv + (long long)blockIdx.z * cBatch;
#pragma unroll
        for (int v = 0; v < 8; v++)
          C[(long long)(mBase + v) * ldc + nCol] = c[v];
      }
    }
  }
}

// WT[n,k] = (bf16) W[k,n]
__global__ __launch_bounds__(256)
void transpose_convert(const float* __restrict__ W, bf16* __restrict__ WT, int Dm) {
  __shared__ float t[32][33];
  const int tx = threadIdx.x & 31;
  const int ty = threadIdx.x >> 5;
  const int bx = blockIdx.x * 32;
  const int by = blockIdx.y * 32;
#pragma unroll
  for (int j = 0; j < 32; j += 8)
    t[ty + j][tx] = W[(long long)(by + ty + j) * Dm + bx + tx];
  __syncthreads();
#pragma unroll
  for (int j = 0; j < 32; j += 8)
    WT[(long long)(bx + ty + j) * Dm + by + tx] = (bf16)t[tx][ty + j];
}

__global__ __launch_bounds__(256)
void convert_bf16(const float* __restrict__ x, bf16* __restrict__ y, int n) {
  const int i = (blockIdx.x * 256 + threadIdx.x) * 4;
  if (i + 3 < n) {
    v4f in = *(const v4f*)(x + i);
    v4bf o;
#pragma unroll
    for (int k = 0; k < 4; k++) o[k] = (bf16)in[k];
    *(v4bf*)(y + i) = o;
  }
}

// in-place row softmax over 2048 bf16 cols; one block per row
__global__ __launch_bounds__(256)
void softmax_rows(bf16* __restrict__ s, int ncols) {
  bf16* row = s + (long long)blockIdx.x * ncols;
  __shared__ float red[256];
  const int tid = threadIdx.x;
  float v[8];
#pragma unroll
  for (int i = 0; i < 8; i++) v[i] = (float)row[tid + i * 256];
  float mx = v[0];
#pragma unroll
  for (int i = 1; i < 8; i++) mx = fmaxf(mx, v[i]);
  red[tid] = mx; __syncthreads();
  for (int k = 128; k > 0; k >>= 1) {
    if (tid < k) red[tid] = fmaxf(red[tid], red[tid + k]);
    __syncthreads();
  }
  mx = red[0]; __syncthreads();
  float sum = 0.f;
#pragma unroll
  for (int i = 0; i < 8; i++) { v[i] = __expf(v[i] - mx); sum += v[i]; }
  red[tid] = sum; __syncthreads();
  for (int k = 128; k > 0; k >>= 1) {
    if (tid < k) red[tid] += red[tid + k];
    __syncthreads();
  }
  const float inv = 1.0f / red[0];
#pragma unroll
  for (int i = 0; i < 8; i++) row[tid + i * 256] = (bf16)(v[i] * inv);
}

// x = LN(yin + resid)*g + b ; writes f32 x and bf16 copy. One block per row (2048 cols).
__global__ __launch_bounds__(256)
void residual_layernorm(const float* __restrict__ yin, const float* __restrict__ resid,
                        const float* __restrict__ g, const float* __restrict__ b,
                        float* __restrict__ xout, bf16* __restrict__ xbfo, int ncols) {
  const long long base = (long long)blockIdx.x * ncols;
  __shared__ float r1[256], r2[256];
  const int tid = threadIdx.x;
  float v[8]; float s = 0.f, q = 0.f;
#pragma unroll
  for (int i = 0; i < 8; i++) {
    const int c = tid + i * 256;
    const float t = yin[base + c] + resid[base + c];
    v[i] = t; s += t; q += t * t;
  }
  r1[tid] = s; r2[tid] = q; __syncthreads();
  for (int k = 128; k > 0; k >>= 1) {
    if (tid < k) { r1[tid] += r1[tid + k]; r2[tid] += r2[tid + k]; }
    __syncthreads();
  }
  const float invn = 1.0f / (float)ncols;
  const float mu  = r1[0] * invn;
  const float var = r2[0] * invn - mu * mu;
  const float rs  = rsqrtf(var + 1e-5f);
#pragma unroll
  for (int i = 0; i < 8; i++) {
    const int c = tid + i * 256;
    const float o = (v[i] - mu) * rs * g[c] + b[c];
    xout[base + c] = o;
    xbfo[base + c] = (bf16)o;
  }
}

extern "C" void kernel_launch(void* const* d_in, const int* in_sizes, int n_in,
                              void* d_out, int out_size, void* d_ws, size_t ws_size,
                              hipStream_t stream)
{
  (void)in_sizes; (void)n_in; (void)out_size; (void)ws_size;
  const int S = 2048, D = 2048, H = 16, DK = 128;
  const long long SD = (long long)S * D;
  const float* Xin = (const float*)d_in[0];

  char* base = (char*)d_ws;
  size_t off = 0;
  auto take = [&](size_t bytes) -> void* {
    void* r = base + off;
    off = (off + bytes + 255) & ~(size_t)255;
    return r;
  };
  float* xcur   = (float*)take(SD * 4);
  float* fcout  = (float*)take(SD * 4);
  bf16*  xbf    = (bf16*) take(SD * 2);
  bf16*  qbf    = (bf16*) take(SD * 2);
  bf16*  kbf    = (bf16*) take(SD * 2);
  bf16*  vt     = (bf16*) take(SD * 2);   // V^T (D x S), feature-major
  bf16*  ctxT   = (bf16*) take(SD * 2);   // scrambled ctx (= transposed ctx)
  bf16*  wbuf   = (bf16*) take(SD * 2);   // one transposed weight at a time
  bf16*  scores = (bf16*) take((size_t)H * S * S * 2);  // also holds probs in-place

  dim3 blk(256);
  dim3 gFull(D / 128, S / 128, 1);
  dim3 gScore(S / 128, S / 128, H);
  dim3 gCtx(DK / 128, S / 128, H);
  dim3 gT(D / 32, D / 32, 1);
  const float scl = 0.088388347648318447f;  // 1/sqrt(128)

  convert_bf16<<<dim3((S * D) / (256 * 4)), blk, 0, stream>>>(Xin, xbf, S * D);

  auto tw = [&](const float* W) {
    transpose_convert<<<gT, blk, 0, stream>>>(W, wbuf, D);
  };

  auto layer = [&](const float* resid0,
                   const float* wq, const float* wk, const float* wv,
                   const float* wfc, const float* bfc,
                   const float* g1, const float* b1,
                   const float* wl, const float* bl,
                   const float* g2, const float* b2,
                   float* xfinal, bool last)
  {
    // Q = X wq ; K = X wk (row-major bf16) ; V^T (transposed-store bf16)
    tw(wq);
    gemm_wmma<EPI_BF16N><<<gFull, blk, 0, stream>>>(xbf, wbuf, qbf, nullptr,
        D, D, D, D, 0, 0, 0, 1.0f);
    tw(wk);
    gemm_wmma<EPI_BF16N><<<gFull, blk, 0, stream>>>(xbf, wbuf, kbf, nullptr,
        D, D, D, D, 0, 0, 0, 1.0f);
    tw(wv);
    gemm_wmma<EPI_BF16T><<<gFull, blk, 0, stream>>>(xbf, wbuf, vt, nullptr,
        D, D, D, S, 0, 0, 0, 1.0f);
    // scores_h = (Q_h K_h^T) * scale  (batched over heads; BT operand = K_h row-major)
    gemm_wmma<EPI_BF16N | EPI_SCALE><<<gScore, blk, 0, stream>>>(qbf, kbf, scores, nullptr,
        DK, D, D, S, (long long)DK, (long long)DK, (long long)S * S, scl);
    // softmax over keys, in place
    softmax_rows<<<dim3(H * S), blk, 0, stream>>>(scores, S);
    // ctx^T (the reference's scramble) = (P_h V_h)^T via transposed-store epilogue
    gemm_wmma<EPI_BF16T><<<gCtx, blk, 0, stream>>>(scores, vt, ctxT, nullptr,
        S, S, S, S, (long long)S * S, (long long)DK * S, (long long)DK * S, 1.0f);
    // fc: scrambled @ wfc + bfc  (f32 out)
    tw(wfc);
    gemm_wmma<EPI_BIAS><<<gFull, blk, 0, stream>>>(ctxT, wbuf, fcout, bfc,
        D, D, D, D, 0, 0, 0, 1.0f);
    residual_layernorm<<<dim3(S), blk, 0, stream>>>(fcout, resid0, g1, b1, xcur, xbf, D);
    // linear: X @ wl + bl, then LN
    tw(wl);
    gemm_wmma<EPI_BIAS><<<gFull, blk, 0, stream>>>(xbf, wbuf, fcout, bl,
        D, D, D, D, 0, 0, 0, 1.0f);
    residual_layernorm<<<dim3(S), blk, 0, stream>>>(fcout, xcur, g2, b2,
        last ? xfinal : xcur, xbf, D);
  };

  layer(Xin,
        (const float*)d_in[1],  (const float*)d_in[2],  (const float*)d_in[3],
        (const float*)d_in[4],  (const float*)d_in[5],
        (const float*)d_in[6],  (const float*)d_in[7],
        (const float*)d_in[8],  (const float*)d_in[9],
        (const float*)d_in[10], (const float*)d_in[11],
        nullptr, false);
  layer(xcur,
        (const float*)d_in[12], (const float*)d_in[13], (const float*)d_in[14],
        (const float*)d_in[15], (const float*)d_in[16],
        (const float*)d_in[17], (const float*)d_in[18],
        (const float*)d_in[19], (const float*)d_in[20],
        (const float*)d_in[21], (const float*)d_in[22],
        (float*)d_out, true);
}
